// EDM_task1_85212151153315
// MI455X (gfx1250) — matrix operand order repaired and checked by reading, not compiled, not run on hardware
//
#include <hip/hip_runtime.h>
#include <hip/hip_bf16.h>

typedef __attribute__((ext_vector_type(16))) __bf16 v16bf;
typedef __attribute__((ext_vector_type(8)))  __bf16 v8bf;
typedef __attribute__((ext_vector_type(8)))  float  v8f;

#define DK          2048
#define BATCH       64
#define MAXOBJS     128
#define NOBJ        200
#define NACT        100
#define NCLS        2000
#define NFLAT       (MAXOBJS * NACT)   // 12800
#define KSTEP       64                 // K per pipeline stage
#define NSTAGE      (DK / KSTEP)       // 32
#define LSTRIDE     72                 // padded LDS row stride (elements)

// order-preserving float <-> uint encoding for atomicMax-based segment max
__device__ __forceinline__ unsigned fenc(float f) {
  unsigned u = __float_as_uint(f);
  return (u & 0x80000000u) ? ~u : (u | 0x80000000u);
}
__device__ __forceinline__ float fdec(unsigned u) {
  return (u & 0x80000000u) ? __uint_as_float(u & 0x7FFFFFFFu)
                           : __uint_as_float(~u);
}

#define WMMA_BF16(ACC, AF, BF) \
  ACC = __builtin_amdgcn_wmma_f32_16x16x32_bf16(false, AF, false, BF, \
                                                (short)0, ACC, false, false)

// ---------------------------------------------------------------------------
// bulk f32 -> bf16 downconvert (8 elements / thread, b128 in, b128 out)
// ---------------------------------------------------------------------------
__global__ void __launch_bounds__(256)
cvt_kernel(const float* __restrict__ src, __bf16* __restrict__ dst, int n8) {
  const int i = blockIdx.x * 256 + threadIdx.x;
  if (i >= n8) return;
  float4 a = ((const float4*)src)[2 * i + 0];
  float4 b = ((const float4*)src)[2 * i + 1];
  v8bf o;
  o[0] = (__bf16)a.x; o[1] = (__bf16)a.y; o[2] = (__bf16)a.z; o[3] = (__bf16)a.w;
  o[4] = (__bf16)b.x; o[5] = (__bf16)b.y; o[6] = (__bf16)b.z; o[7] = (__bf16)b.w;
  ((v8bf*)dst)[i] = o;
}

// ---------------------------------------------------------------------------
// masked mean pooling over objects -> bf16 pooled[b][d]
// ---------------------------------------------------------------------------
__global__ void __launch_bounds__(256)
pool_kernel(const float* __restrict__ inp, const float* __restrict__ objmask,
            __bf16* __restrict__ pooledb) {
  const int b   = blockIdx.x;
  const int tid = threadIdx.x;
  const float* mrow = objmask + b * MAXOBJS;
  float msum = 0.f;
  #pragma unroll 8
  for (int m = 0; m < MAXOBJS; ++m) msum += mrow[m];
  const float inv = 1.0f / msum;
  const float* base = inp + (size_t)b * MAXOBJS * DK;
  for (int d = tid; d < DK; d += 256) {
    float acc = 0.f;
    #pragma unroll 4
    for (int m = 0; m < MAXOBJS; ++m)
      acc += base[(size_t)m * DK + d] * mrow[m];
    pooledb[b * DK + d] = (__bf16)(acc * inv);
  }
}

// ---------------------------------------------------------------------------
// out[row][col] = A[row][:] . W[col][:] + bias[col]     (A, W in bf16)
// Wave tile 16x64 (4 f32 accumulators). Block = 8 waves stacked in M
// -> 128x64 block tile. BOTH A and B staged in LDS via async-to-LDS copies,
// double buffered. Stage loop is kept rolled (unroll 1) so accumulators stay
// in fixed registers; all fragments are loaded before the 8 back-to-back
// WMMAs of a stage.
// ---------------------------------------------------------------------------
__global__ void __launch_bounds__(256)
gemm_bf16_wmma(const __bf16* __restrict__ A, const __bf16* __restrict__ W,
               const float* __restrict__ bias, float* __restrict__ out,
               int Mrows, int Nout) {
  __shared__ __align__(32) __bf16 ldsA[2 * 128 * LSTRIDE];  // 2 x 18 KB
  __shared__ __align__(32) __bf16 ldsB[2 * 64 * LSTRIDE];   // 2 x  9 KB

  const int tid  = threadIdx.x;
  const int lane = tid & 31;
  const int wave = tid >> 5;
  const int half = lane >> 4;        // K-half selector inside fragments
  const int idx  = lane & 15;        // M (A) / N (B,C) within 16x16 tile

  const int rowBase = blockIdx.x * 128 + wave * 16;
  const int colBase = blockIdx.y * 64;

  // ---- cooperative async A copy: thread -> (row, 64B half-row) -----------
  const int acp_row  = tid >> 1;               // 0..127
  const int acp_koff = (tid & 1) * 32;         // elements: 0 or 32
  int acp_grow = blockIdx.x * 128 + acp_row;
  if (acp_grow >= Mrows) acp_grow = Mrows - 1;               // clamp tail
  const __bf16* aSrc = A + (size_t)acp_grow * DK + acp_koff;
  const unsigned aLds = (unsigned)(uintptr_t)(&ldsA[0]) +
                        (unsigned)((acp_row * LSTRIDE + acp_koff) * 2);
  const unsigned aBufStride = 128u * LSTRIDE * 2u;

  // ---- cooperative async B copy: thread -> (col, 32B K chunk) ------------
  const int bcp_col  = tid >> 2;               // 0..63
  const int bcp_koff = (tid & 3) * 16;         // elements: 0,16,32,48
  int bcp_gcol = colBase + bcp_col;
  if (bcp_gcol >= Nout) bcp_gcol = Nout - 1;   // clamp load, guard store
  const __bf16* bSrc = W + (size_t)bcp_gcol * DK + bcp_koff;
  const unsigned bLds = (unsigned)(uintptr_t)(&ldsB[0]) +
                        (unsigned)((bcp_col * LSTRIDE + bcp_koff) * 2);
  const unsigned bBufStride = 64u * LSTRIDE * 2u;

  auto issueStage = [&](int s, int buf) {
    const __bf16* ga = aSrc + (size_t)s * KSTEP;           // 64 B / thread
    const unsigned la = aLds + (unsigned)buf * aBufStride;
    asm volatile("global_load_async_to_lds_b128 %0, %1, off"
                 :: "v"(la), "v"(ga) : "memory");
    asm volatile("global_load_async_to_lds_b128 %0, %1, off offset:16"
                 :: "v"(la), "v"(ga) : "memory");
    asm volatile("global_load_async_to_lds_b128 %0, %1, off offset:32"
                 :: "v"(la), "v"(ga) : "memory");
    asm volatile("global_load_async_to_lds_b128 %0, %1, off offset:48"
                 :: "v"(la), "v"(ga) : "memory");
    const __bf16* gb = bSrc + (size_t)s * KSTEP;           // 32 B / thread
    const unsigned lb = bLds + (unsigned)buf * bBufStride;
    asm volatile("global_load_async_to_lds_b128 %0, %1, off"
                 :: "v"(lb), "v"(gb) : "memory");
    asm volatile("global_load_async_to_lds_b128 %0, %1, off offset:16"
                 :: "v"(lb), "v"(gb) : "memory");
  };

  v8f acc0{}, acc1{}, acc2{}, acc3{};

  // per-wave fragment base offsets inside one LDS buffer
  const int aFragOff = (wave * 16 + idx) * LSTRIDE + half * 16;
  const int bFragOff = idx * LSTRIDE + half * 16;

  auto computeStage = [&](int buf) {
    const __bf16* Ab = ldsA + buf * (128 * LSTRIDE) + aFragOff;
    const __bf16* Bb = ldsB + buf * (64 * LSTRIDE) + bFragOff;
    // ---- load ALL fragments for this stage first -------------------------
    v8bf a00 = *(const v8bf*)(Ab + 0);       // kk=0 chunks
    v8bf a01 = *(const v8bf*)(Ab + 8);
    v8bf a10 = *(const v8bf*)(Ab + 32);      // kk=1 chunks
    v8bf a11 = *(const v8bf*)(Ab + 40);
    v16bf b00 = *(const v16bf*)(Bb + 0 * 16 * LSTRIDE);
    v16bf b01 = *(const v16bf*)(Bb + 1 * 16 * LSTRIDE);
    v16bf b02 = *(const v16bf*)(Bb + 2 * 16 * LSTRIDE);
    v16bf b03 = *(const v16bf*)(Bb + 3 * 16 * LSTRIDE);
    v16bf b10 = *(const v16bf*)(Bb + 0 * 16 * LSTRIDE + 32);
    v16bf b11 = *(const v16bf*)(Bb + 1 * 16 * LSTRIDE + 32);
    v16bf b12 = *(const v16bf*)(Bb + 2 * 16 * LSTRIDE + 32);
    v16bf b13 = *(const v16bf*)(Bb + 3 * 16 * LSTRIDE + 32);
    v16bf af0 = __builtin_shufflevector(
        a00, a01, 0, 1, 2, 3, 4, 5, 6, 7, 8, 9, 10, 11, 12, 13, 14, 15);
    v16bf af1 = __builtin_shufflevector(
        a10, a11, 0, 1, 2, 3, 4, 5, 6, 7, 8, 9, 10, 11, 12, 13, 14, 15);
    // ---- 8 back-to-back WMMAs -------------------------------------------
    WMMA_BF16(acc0, af0, b00);
    WMMA_BF16(acc1, af0, b01);
    WMMA_BF16(acc2, af0, b02);
    WMMA_BF16(acc3, af0, b03);
    WMMA_BF16(acc0, af1, b10);
    WMMA_BF16(acc1, af1, b11);
    WMMA_BF16(acc2, af1, b12);
    WMMA_BF16(acc3, af1, b13);
  };

  // ---- prologue ----------------------------------------------------------
  int buf = 0;
  issueStage(0, 0);
  asm volatile("s_wait_asynccnt 0x0" ::: "memory");
  __syncthreads();

  // ---- rolled steady-state loop (last stage peeled) ----------------------
  #pragma unroll 1
  for (int s = 0; s < NSTAGE - 1; ++s) {
    issueStage(s + 1, buf ^ 1);        // fill other buffer during compute
    computeStage(buf);                 // 8 x v_wmma, accumulate in place
    asm volatile("s_wait_asynccnt 0x0" ::: "memory");
    __syncthreads();
    buf ^= 1;
  }
  computeStage(buf);

  // ---- store: lane idx = N, VGPR r -> M = r + 8*half ---------------------
  #pragma unroll
  for (int t = 0; t < 4; ++t) {
    const int c = colBase + t * 16 + idx;
    if (c >= Nout) continue;
    const float bv = bias[c];
    const v8f av = (t == 0) ? acc0 : (t == 1) ? acc1 : (t == 2) ? acc2 : acc3;
    #pragma unroll
    for (int r = 0; r < 8; ++r) {
      const int row = rowBase + r + half * 8;
      if (row < Mrows)
        out[(size_t)row * Nout + c] = av[r] + bv;
    }
  }
}

// ---------------------------------------------------------------------------
// scatter-max of selected action logits into classes + final combine
// ---------------------------------------------------------------------------
__global__ void __launch_bounds__(256)
finalize_kernel(const float* __restrict__ act_out,   // [B, M, A] flat
                const float* __restrict__ objmask,   // [B, M]
                const int*   __restrict__ tgts,      // [B, M*A]
                const float* __restrict__ ne_out,    // [B, C]
                float* __restrict__ final_out) {     // [B, C]
  __shared__ unsigned smax[NCLS];
  __shared__ int      scnt[NCLS];
  const int b   = blockIdx.x;
  const int tid = threadIdx.x;

  for (int c = tid; c < NCLS; c += 256) { smax[c] = 0u; scnt[c] = 0; }
  __syncthreads();

  const float* arow = act_out + (size_t)b * NFLAT;
  const int*   trow = tgts    + (size_t)b * NFLAT;
  const float* mrow = objmask + b * MAXOBJS;

  for (int i = tid; i < NFLAT; i += 256) {
    const float v = arow[i];
    const int   m = i / NACT;
    const int   t = trow[i];
    // sigmoid(v) > 0.5  <=>  v > 0
    const bool sel = (v > 0.0f) && (mrow[m] > 0.0f) && (t >= 0) && (t < NCLS);
    if (sel) {
      atomicMax(&smax[t], fenc(v));
      atomicAdd(&scnt[t], 1);
    }
  }
  __syncthreads();

  for (int c = tid; c < NCLS; c += 256) {
    const int   cnt = scnt[c];
    const float seg = cnt ? fdec(smax[c]) : -__builtin_inff();
    const float mapped = (cnt == NFLAT) ? seg : fmaxf(seg, 0.0f);
    const float o = (mapped != 0.0f) ? mapped : ne_out[(size_t)b * NCLS + c];
    final_out[(size_t)b * NCLS + c] = o;
  }
}

// ---------------------------------------------------------------------------
extern "C" void kernel_launch(void* const* d_in, const int* in_sizes, int n_in,
                              void* d_out, int out_size, void* d_ws, size_t ws_size,
                              hipStream_t stream) {
  const float* inp     = (const float*)d_in[0];
  const float* objmask = (const float*)d_in[1];
  const int*   tgts    = (const int*)  d_in[2];
  const float* W_obj   = (const float*)d_in[3];
  const float* b_obj   = (const float*)d_in[4];
  const float* W_act   = (const float*)d_in[5];
  const float* b_act   = (const float*)d_in[6];
  const float* W_ne    = (const float*)d_in[7];
  const float* b_ne    = (const float*)d_in[8];

  float* out       = (float*)d_out;
  float* final_out = out;                                      // [64,2000]
  float* act_out   = out + BATCH * NCLS;                       // [64,128,100]
  float* obj_out   = act_out + (size_t)BATCH * MAXOBJS * NACT; // [64,128,200]

  const int Mrows = BATCH * MAXOBJS;                           // 8192

  // workspace layout (bf16 mirrors + ne_out scratch), ~44 MB
  char* ws = (char*)d_ws;
  __bf16* inpB  = (__bf16*)ws;                ws += (size_t)Mrows * DK * 2;
  __bf16* WactB = (__bf16*)ws;                ws += (size_t)NACT * DK * 2;
  __bf16* WobjB = (__bf16*)ws;                ws += (size_t)NOBJ * DK * 2;
  __bf16* WneB  = (__bf16*)ws;                ws += (size_t)NCLS * DK * 2;
  __bf16* poolB = (__bf16*)ws;                ws += (size_t)BATCH * DK * 2;
  float*  ne_out = (float*)ws;

  // 1) downconvert operands to bf16 (one pass; everything stays in L2)
  {
    int n8 = Mrows * DK / 8;
    cvt_kernel<<<(n8 + 255) / 256, 256, 0, stream>>>(inp, inpB, n8);
    n8 = NACT * DK / 8;
    cvt_kernel<<<(n8 + 255) / 256, 256, 0, stream>>>(W_act, WactB, n8);
    n8 = NOBJ * DK / 8;
    cvt_kernel<<<(n8 + 255) / 256, 256, 0, stream>>>(W_obj, WobjB, n8);
    n8 = NCLS * DK / 8;
    cvt_kernel<<<(n8 + 255) / 256, 256, 0, stream>>>(W_ne, WneB, n8);
  }

  // 2) masked mean pool (f32 math, bf16 result for the ne GEMM)
  pool_kernel<<<BATCH, 256, 0, stream>>>(inp, objmask, poolB);

  // 3) WMMA GEMMs
  dim3 gAct(Mrows / 128, (NACT + 63) / 64);
  gemm_bf16_wmma<<<gAct, 256, 0, stream>>>(inpB, WactB, b_act, act_out,
                                           Mrows, NACT);
  dim3 gObj(Mrows / 128, (NOBJ + 63) / 64);
  gemm_bf16_wmma<<<gObj, 256, 0, stream>>>(inpB, WobjB, b_obj, obj_out,
                                           Mrows, NOBJ);
  dim3 gNe(1, (NCLS + 63) / 64);
  gemm_bf16_wmma<<<gNe, 256, 0, stream>>>(poolB, WneB, b_ne, ne_out,
                                          BATCH, NCLS);

  // 4) scatter-max + combine
  finalize_kernel<<<BATCH, 256, 0, stream>>>(act_out, objmask, tgts,
                                             ne_out, final_out);
}
